// CharToWord_79121887526915
// MI455X (gfx1250) — matrix operand order repaired and checked by reading, not compiled, not run on hardware
//
#include <hip/hip_runtime.h>
#include <hip/hip_bf16.h>

// ---------------- problem constants ----------------
#define B_   16
#define L_   2048
#define D_   128
#define W_   512
#define C_   512
#define F_   1664      // sum of conv filters
#define C2_  1024
#define BW_  (B_ * W_) // 8192
#define PAD_ 16        // L-padding rows (halo for conv shifts, max |shift| = 4)
#define LP_  (L_ + 2 * PAD_)
#define EPS_ 1e-5f

typedef __attribute__((ext_vector_type(16))) _Float16 v16h;
typedef __attribute__((ext_vector_type(8)))  _Float16 v8h;
typedef __attribute__((ext_vector_type(8)))  float    v8f;

// =====================================================================
// WMMA fragment helpers (CDNA5 16x16x32 f16 layouts, ISA 05_wmma §7.12.2)
//
// A (16x32 MxK, row-major, unit k-stride): lane m = m0+(lane&15),
//   lanes 0-15 hold K = {0..7, 16..23}, lanes 16-31 hold K = {8..15, 24..31}
//   -> two contiguous 16B chunks at k0+kb and k0+kb+16 (kb = 0 or 8).
// B (32x16 KxN) stored TRANSPOSED (N-major, ldk = K): lane n fixed,
//   lanes 0-15 hold K = {0..7, 16..23}, lanes 16-31 hold K = {8..15, 24..31}
//   -> two contiguous 16B chunks at k0+kb and k0+kb+16 (kb = 0 or 8).
// C/D f32: vgpr r -> row m0 + r + (lane<16 ? 0 : 8), col n0 + (lane&15)
// =====================================================================
__device__ __forceinline__ v16h cat16(v8h lo, v8h hi) {
  return __builtin_shufflevector(lo, hi, 0, 1, 2, 3, 4, 5, 6, 7,
                                 8, 9, 10, 11, 12, 13, 14, 15);
}

__device__ __forceinline__ v16h frag_a(const _Float16* __restrict__ A, int lda,
                                       int m0, int k0, int lane) {
  const int m  = m0 + (lane & 15);
  const int kb = (lane & 16) ? 8 : 0;
  const _Float16* p = A + (size_t)m * lda + (size_t)(k0 + kb);
  v8h lo = *(const v8h*)p;
  v8h hi = *(const v8h*)(p + 16);
  return cat16(lo, hi);
}

// BT is the transposed B operand, row n, ldk = K dimension
__device__ __forceinline__ v16h frag_bt(const _Float16* __restrict__ BT, int ldk,
                                        int n, int k0, int lane) {
  const int kg = (lane & 16) ? 8 : 0;
  const _Float16* p = BT + (size_t)n * ldk + (size_t)(k0 + kg);
  v8h lo = *(const v8h*)p;
  v8h hi = *(const v8h*)(p + 16);
  return cat16(lo, hi);
}

__device__ __forceinline__ v8f wmma32(v16h a, v16h b, v8f c) {
  return __builtin_amdgcn_wmma_f32_16x16x32_f16(false, a, false, b, (short)0, c,
                                                false, false);
}

__device__ __forceinline__ v8f vzero8() {
  v8f z;
#pragma unroll
  for (int r = 0; r < 8; ++r) z[r] = 0.0f;
  return z;
}

// =====================================================================
// Utility kernels
// =====================================================================
__global__ void k_f32_to_f16(_Float16* __restrict__ dst, const float* __restrict__ src,
                             int n) {
  int i = blockIdx.x * blockDim.x + threadIdx.x;
  if (i < n) dst[i] = (_Float16)src[i];
}

// transpose-convert: src f32 (R x Cc) -> dst f16 (Cc x R)
__global__ void k_cvt_t(_Float16* __restrict__ dst, const float* __restrict__ src,
                        int R, int Cc) {
  int i = blockIdx.x * blockDim.x + threadIdx.x;
  if (i >= R * Cc) return;
  int r = i % R, c = i / R;                 // output index i = c*R + r
  dst[i] = (_Float16)src[(size_t)r * Cc + c];
}

// conv weights (f, D, ks) -> packed (ks, f, D) so taps have unit k-stride
__global__ void k_pack_conv(_Float16* __restrict__ dst, const float* __restrict__ src,
                            int fcnt, int ks) {
  int i = blockIdx.x * blockDim.x + threadIdx.x;
  if (i >= fcnt * D_ * ks) return;
  int d = i % D_;
  int f = (i / D_) % fcnt;
  int t = i / (D_ * fcnt);
  dst[i] = (_Float16)src[((size_t)f * D_ + d) * ks + t];
}

__global__ void k_zero_f32(float* __restrict__ p, int n) {
  int i = blockIdx.x * blockDim.x + threadIdx.x;
  if (i < n) p[i] = 0.0f;
}

// x (B,L,D) f32 * mask(B,L) -> x16p (B, PAD+L+PAD, D) f16 with zero halos
__global__ void k_prep_x(const float* __restrict__ emb, const float* __restrict__ mask,
                         _Float16* __restrict__ x16p) {
  int i = blockIdx.x * blockDim.x + threadIdx.x;
  if (i >= B_ * LP_ * D_) return;
  int d  = i % D_;
  int lp = (i / D_) % LP_;
  int b  = i / (D_ * LP_);
  int l  = lp - PAD_;
  float v = 0.0f;
  if (l >= 0 && l < L_)
    v = emb[((size_t)b * L_ + l) * D_ + d] * mask[(size_t)b * L_ + l];
  x16p[i] = (_Float16)v;
}

// per-batch word segmentation: seg[b][l] = searchsorted_right(cumsum(clip(wl,0)), l)
__global__ void k_seg(const int* __restrict__ wl, int* __restrict__ seg) {
  __shared__ int cs[W_];
  const int b = blockIdx.x;
  if (threadIdx.x == 0) {
    int s = 0;
    for (int j = 0; j < W_; ++j) {
      int v = wl[(size_t)b * W_ + j];
      if (v < 0) v = 0;
      s += v;
      cs[j] = s;
    }
  }
  __syncthreads();
  for (int l = threadIdx.x; l < L_; l += blockDim.x) {
    int lo = 0, hi = W_;
    while (lo < hi) {
      int mid = (lo + hi) >> 1;
      if (cs[mid] <= l) lo = mid + 1; else hi = mid;
    }
    seg[(size_t)b * L_ + l] = lo;           // may be == W_ (dropped)
  }
}

// =====================================================================
// Conv bank -> h16 stored (B, L, F) f16 (transposed: ready as BT for after_w)
// Wave tile: 16 rows (one filter) x 32 cols; padded x => NO bounds guards.
// =====================================================================
__global__ void k_conv(const _Float16* __restrict__ x16p,
                       const _Float16* __restrict__ w0, const _Float16* __restrict__ w1,
                       const _Float16* __restrict__ w2, const _Float16* __restrict__ w3,
                       const _Float16* __restrict__ w4,
                       const float* __restrict__ cb0, const float* __restrict__ cb1,
                       const float* __restrict__ cb2, const float* __restrict__ cb3,
                       const float* __restrict__ cb4,
                       _Float16* __restrict__ h16) {
  const int lane = threadIdx.x & 31;
  const int wv   = blockIdx.x * (blockDim.x >> 5) + (threadIdx.x >> 5);
  const int CT = F_ / 16, NT = L_ / 32;      // 104, 64
  if (wv >= B_ * CT * NT) return;
  const int nt = wv % NT;
  const int ct = (wv / NT) % CT;
  const int b  = wv / (NT * CT);
  const int cg = ct * 16;

  int fi, foff, fcnt, ks;
  const _Float16* Wt; const float* Bs;
  if      (cg < 128)  { fi = 0; foff = 0;    fcnt = 128; ks = 1; Wt = w0; Bs = cb0; }
  else if (cg < 384)  { fi = 1; foff = 128;  fcnt = 256; ks = 3; Wt = w1; Bs = cb1; }
  else if (cg < 896)  { fi = 2; foff = 384;  fcnt = 512; ks = 5; Wt = w2; Bs = cb2; }
  else if (cg < 1408) { fi = 3; foff = 896;  fcnt = 512; ks = 7; Wt = w3; Bs = cb3; }
  else                { fi = 4; foff = 1408; fcnt = 256; ks = 9; Wt = w4; Bs = cb4; }

  const int m0 = cg - foff;
  const int n0 = nt * 32;
  // padded (LP_, D) per batch; interior row l lives at (l + PAD_)
  const _Float16* xb = x16p + ((size_t)b * LP_ + PAD_) * D_;

  v8f acc0 = vzero8(), acc1 = vzero8();

  for (int t = 0; t < ks; ++t) {
    const _Float16* At = Wt + (size_t)t * fcnt * D_;   // packed (ks, f, D)
    const int col0 = n0 + (lane & 15) + t - fi;        // pad = fi; in [-4, L+4)
#pragma unroll
    for (int k0 = 0; k0 < D_; k0 += 32) {
      v16h af  = frag_a(At, D_, m0, k0, lane);
      v16h bf0 = frag_bt(xb, D_, col0,      k0, lane);
      v16h bf1 = frag_bt(xb, D_, col0 + 16, k0, lane);
      acc0 = wmma32(af, bf0, acc0);
      acc1 = wmma32(af, bf1, acc1);
    }
  }

  // epilogue: 8 consecutive channels per lane -> one b128 store per half
  const int mbase = m0 + ((lane & 16) ? 8 : 0);
  float bias[8];
#pragma unroll
  for (int r = 0; r < 8; ++r) bias[r] = Bs[mbase + r];

#pragma unroll
  for (int half = 0; half < 2; ++half) {
    const v8f acc = half ? acc1 : acc0;
    const int l = n0 + (lane & 15) + half * 16;
    v8h o;
#pragma unroll
    for (int r = 0; r < 8; ++r) o[r] = (_Float16)(acc[r] + bias[r]);
    *(v8h*)(h16 + ((size_t)b * L_ + l) * F_ + foff + mbase) = o;
  }
}

// =====================================================================
// after_w (512 x 1664) GEMM + ReLU, fused segment-max pool into
// shrinked (B, W, C) via int-ordered f32 atomic max (values >= 0).
// Wave tile: 32x32 (4 accumulators), K = 1664.
// =====================================================================
__global__ void k_after(const _Float16* __restrict__ h16, const _Float16* __restrict__ aw,
                        const float* __restrict__ ab, const int* __restrict__ seg,
                        float* __restrict__ shr) {
  const int lane = threadIdx.x & 31;
  const int wv   = blockIdx.x * (blockDim.x >> 5) + (threadIdx.x >> 5);
  const int MT = C_ / 32, NT = L_ / 32;      // 16, 64
  if (wv >= B_ * MT * NT) return;
  const int nt = wv % NT;
  const int mt = (wv / NT) % MT;
  const int b  = wv / (NT * MT);
  const int m0 = mt * 32, n0 = nt * 32;
  const _Float16* hb = h16 + (size_t)b * L_ * F_;   // (L, F): BT with ldk = F
  const int nA = n0 + (lane & 15), nB = nA + 16;

  v8f a00 = vzero8(), a01 = vzero8(), a10 = vzero8(), a11 = vzero8();

#pragma unroll 2
  for (int k0 = 0; k0 < F_; k0 += 32) {
    __builtin_prefetch(hb + (size_t)nA * F_ + k0 + 128, 0, 1);
    v16h af0 = frag_a(aw, F_, m0,      k0, lane);
    v16h af1 = frag_a(aw, F_, m0 + 16, k0, lane);
    v16h bf0 = frag_bt(hb, F_, nA, k0, lane);
    v16h bf1 = frag_bt(hb, F_, nB, k0, lane);
    a00 = wmma32(af0, bf0, a00);
    a01 = wmma32(af0, bf1, a01);
    a10 = wmma32(af1, bf0, a10);
    a11 = wmma32(af1, bf1, a11);
  }

  const int w0 = seg[(size_t)b * L_ + nA];
  const int w1 = seg[(size_t)b * L_ + nB];
#pragma unroll
  for (int q = 0; q < 4; ++q) {
    const v8f acc = (q == 0) ? a00 : (q == 1) ? a01 : (q == 2) ? a10 : a11;
    const int mi = (q >> 1) * 16;
    const int word = (q & 1) ? w1 : w0;
    if (word < W_) {
      float* dst = &shr[((size_t)b * W_ + word) * C_];
#pragma unroll
      for (int r = 0; r < 8; ++r) {
        const int m = m0 + mi + r + ((lane & 16) ? 8 : 0);
        float v = acc[r] + ab[m];
        v = v > 0.0f ? v : 0.0f;
        atomicMax((int*)&dst[m], __float_as_int(v));
      }
    }
  }
}

// =====================================================================
// Highway: pr = relu(Pw@y+pb); g = sigmoid(Tw@y+tb); y' = g*pr+(1-g)*y
// y kept (B, W, C): serves as BT (ldk=C) and gives contiguous epilogue.
// Wave tile: 16(M) x 32(N), P and G accumulators share B fragments.
// =====================================================================
__global__ void k_highway(const _Float16* __restrict__ pw, const float* __restrict__ pb,
                          const _Float16* __restrict__ tw, const float* __restrict__ tb,
                          const float* __restrict__ yin32, const _Float16* __restrict__ yin16,
                          float* __restrict__ yout32, _Float16* __restrict__ yout16) {
  const int lane = threadIdx.x & 31;
  const int wv   = blockIdx.x * (blockDim.x >> 5) + (threadIdx.x >> 5);
  const int MT = C_ / 16, NT = W_ / 32;      // 32, 16
  if (wv >= B_ * MT * NT) return;
  const int nt = wv % NT;
  const int mt = (wv / NT) % MT;
  const int b  = wv / (NT * MT);
  const int m0 = mt * 16, n0 = nt * 32;
  const _Float16* yb = yin16 + (size_t)b * W_ * C_;
  const int nA = n0 + (lane & 15), nB = nA + 16;

  v8f p0 = vzero8(), p1 = vzero8(), g0 = vzero8(), g1 = vzero8();

#pragma unroll 4
  for (int k0 = 0; k0 < C_; k0 += 32) {
    v16h bf0 = frag_bt(yb, C_, nA, k0, lane);
    v16h bf1 = frag_bt(yb, C_, nB, k0, lane);
    v16h ap = frag_a(pw, C_, m0, k0, lane);
    p0 = wmma32(ap, bf0, p0);
    p1 = wmma32(ap, bf1, p1);
    v16h ag = frag_a(tw, C_, m0, k0, lane);
    g0 = wmma32(ag, bf0, g0);
    g1 = wmma32(ag, bf1, g1);
  }

  const int mbase = m0 + ((lane & 16) ? 8 : 0);
#pragma unroll
  for (int half = 0; half < 2; ++half) {
    const v8f accP = half ? p1 : p0;
    const v8f accG = half ? g1 : g0;
    const int n = half ? nB : nA;
    const size_t base = ((size_t)b * W_ + n) * C_ + mbase;
    const v8f yold = *(const v8f*)(yin32 + base);
    v8f o32; v8h o16;
#pragma unroll
    for (int r = 0; r < 8; ++r) {
      const int m = mbase + r;
      float pr = accP[r] + pb[m];
      pr = pr > 0.0f ? pr : 0.0f;
      const float g = 1.0f / (1.0f + expf(-(accG[r] + tb[m])));
      const float yo = g * pr + (1.0f - g) * yold[r];
      o32[r] = yo;
      o16[r] = (_Float16)yo;
    }
    *(v8f*)(yout32 + base) = o32;
    *(v8h*)(yout16 + base) = o16;
  }
}

// =====================================================================
// relu + LayerNorm(ah) over rows of (B, W, C)
// =====================================================================
__global__ void k_relu_ln(const float* __restrict__ yin, const float* __restrict__ g,
                          const float* __restrict__ bt, float* __restrict__ out32,
                          _Float16* __restrict__ out16) {
  __shared__ float sd[256];
  const int row = blockIdx.x;                // (b*W + w)
  const int tid = threadIdx.x;
  const int c0 = 2 * tid, c1 = 2 * tid + 1;
  const size_t base = (size_t)row * C_;

  float v0 = yin[base + c0];
  float v1 = yin[base + c1];
  v0 = v0 > 0.0f ? v0 : 0.0f;
  v1 = v1 > 0.0f ? v1 : 0.0f;

  sd[tid] = v0 + v1;
  __syncthreads();
  for (int s = 128; s > 0; s >>= 1) { if (tid < s) sd[tid] += sd[tid + s]; __syncthreads(); }
  const float mean = sd[0] * (1.0f / C_);
  __syncthreads();
  sd[tid] = (v0 - mean) * (v0 - mean) + (v1 - mean) * (v1 - mean);
  __syncthreads();
  for (int s = 128; s > 0; s >>= 1) { if (tid < s) sd[tid] += sd[tid + s]; __syncthreads(); }
  const float rstd = rsqrtf(sd[0] * (1.0f / C_) + EPS_);

  const float o0 = g[c0] * (v0 - mean) * rstd + bt[c0];
  const float o1 = g[c1] * (v1 - mean) * rstd + bt[c1];
  out32[base + c0] = o0;
  out32[base + c1] = o1;
  out16[base + c0] = (_Float16)o0;
  out16[base + c1] = (_Float16)o1;
}

// =====================================================================
// FF GEMMs (32x32 wave tiles): z1 = relu(Y @ W1 + b1); z = z1 @ W2 + b2
// W1, W2 pre-transposed (N-major) so B fragments are contiguous.
// =====================================================================
__global__ void k_ff1(const _Float16* __restrict__ y16, const _Float16* __restrict__ w1t,
                      const float* __restrict__ b1, _Float16* __restrict__ z1) {
  const int lane = threadIdx.x & 31;
  const int wv   = blockIdx.x * (blockDim.x >> 5) + (threadIdx.x >> 5);
  const int MT = BW_ / 32, NT = C2_ / 32;    // 256, 32
  if (wv >= MT * NT) return;
  const int nt = wv % NT, mt = wv / NT;
  const int m0 = mt * 32, n0 = nt * 32;
  const int nA = n0 + (lane & 15), nB = nA + 16;

  v8f a00 = vzero8(), a01 = vzero8(), a10 = vzero8(), a11 = vzero8();

#pragma unroll 4
  for (int k0 = 0; k0 < C_; k0 += 32) {
    __builtin_prefetch(y16 + (size_t)(m0 + (lane & 15)) * C_ + k0 + 128, 0, 1);
    v16h af0 = frag_a(y16, C_, m0,      k0, lane);
    v16h af1 = frag_a(y16, C_, m0 + 16, k0, lane);
    v16h bf0 = frag_bt(w1t, C_, nA, k0, lane);
    v16h bf1 = frag_bt(w1t, C_, nB, k0, lane);
    a00 = wmma32(af0, bf0, a00);
    a01 = wmma32(af0, bf1, a01);
    a10 = wmma32(af1, bf0, a10);
    a11 = wmma32(af1, bf1, a11);
  }

  const float biasA = b1[nA], biasB = b1[nB];
#pragma unroll
  for (int q = 0; q < 4; ++q) {
    const v8f acc = (q == 0) ? a00 : (q == 1) ? a01 : (q == 2) ? a10 : a11;
    const int mi = (q >> 1) * 16;
    const int n = (q & 1) ? nB : nA;
    const float bias = (q & 1) ? biasB : biasA;
#pragma unroll
    for (int r = 0; r < 8; ++r) {
      const int m = m0 + mi + r + ((lane & 16) ? 8 : 0);
      float v = acc[r] + bias;
      v = v > 0.0f ? v : 0.0f;
      z1[(size_t)m * C2_ + n] = (_Float16)v;
    }
  }
}

__global__ void k_ff2(const _Float16* __restrict__ z1, const _Float16* __restrict__ w2t,
                      const float* __restrict__ b2, float* __restrict__ z) {
  const int lane = threadIdx.x & 31;
  const int wv   = blockIdx.x * (blockDim.x >> 5) + (threadIdx.x >> 5);
  const int MT = BW_ / 32, NT = C_ / 32;     // 256, 16
  if (wv >= MT * NT) return;
  const int nt = wv % NT, mt = wv / NT;
  const int m0 = mt * 32, n0 = nt * 32;
  const int nA = n0 + (lane & 15), nB = nA + 16;

  v8f a00 = vzero8(), a01 = vzero8(), a10 = vzero8(), a11 = vzero8();

#pragma unroll 4
  for (int k0 = 0; k0 < C2_; k0 += 32) {
    __builtin_prefetch(z1 + (size_t)(m0 + (lane & 15)) * C2_ + k0 + 128, 0, 1);
    v16h af0 = frag_a(z1, C2_, m0,      k0, lane);
    v16h af1 = frag_a(z1, C2_, m0 + 16, k0, lane);
    v16h bf0 = frag_bt(w2t, C2_, nA, k0, lane);
    v16h bf1 = frag_bt(w2t, C2_, nB, k0, lane);
    a00 = wmma32(af0, bf0, a00);
    a01 = wmma32(af0, bf1, a01);
    a10 = wmma32(af1, bf0, a10);
    a11 = wmma32(af1, bf1, a11);
  }

  const float biasA = b2[nA], biasB = b2[nB];
#pragma unroll
  for (int q = 0; q < 4; ++q) {
    const v8f acc = (q == 0) ? a00 : (q == 1) ? a01 : (q == 2) ? a10 : a11;
    const int mi = (q >> 1) * 16;
    const int n = (q & 1) ? nB : nA;
    const float bias = (q & 1) ? biasB : biasA;
#pragma unroll
    for (int r = 0; r < 8; ++r) {
      const int m = m0 + mi + r + ((lane & 16) ? 8 : 0);
      z[(size_t)m * C_ + n] = acc[r] + bias;
    }
  }
}

// residual + LayerNorm; optional f16 mirror (null on final layer)
__global__ void k_ln_res(const float* __restrict__ z, const float* __restrict__ yin,
                         const float* __restrict__ g, const float* __restrict__ bt,
                         float* __restrict__ out32, _Float16* __restrict__ out16) {
  __shared__ float sd[256];
  const int row = blockIdx.x;
  const int tid = threadIdx.x;
  const int c0 = 2 * tid, c1 = 2 * tid + 1;
  const size_t base = (size_t)row * C_;

  const float v0 = z[base + c0] + yin[base + c0];
  const float v1 = z[base + c1] + yin[base + c1];

  sd[tid] = v0 + v1;
  __syncthreads();
  for (int s = 128; s > 0; s >>= 1) { if (tid < s) sd[tid] += sd[tid + s]; __syncthreads(); }
  const float mean = sd[0] * (1.0f / C_);
  __syncthreads();
  sd[tid] = (v0 - mean) * (v0 - mean) + (v1 - mean) * (v1 - mean);
  __syncthreads();
  for (int s = 128; s > 0; s >>= 1) { if (tid < s) sd[tid] += sd[tid + s]; __syncthreads(); }
  const float rstd = rsqrtf(sd[0] * (1.0f / C_) + EPS_);

  const float o0 = g[c0] * (v0 - mean) * rstd + bt[c0];
  const float o1 = g[c1] * (v1 - mean) * rstd + bt[c1];
  out32[base + c0] = o0;
  out32[base + c1] = o1;
  if (out16) { out16[base + c0] = (_Float16)o0; out16[base + c1] = (_Float16)o1; }
}

__global__ void k_mask(const int* __restrict__ wl, float* __restrict__ om) {
  int i = blockIdx.x * blockDim.x + threadIdx.x;
  if (i < B_ * W_) om[i] = (wl[i] != -100) ? 1.0f : 0.0f;
}

// =====================================================================
// Launcher
// =====================================================================
extern "C" void kernel_launch(void* const* d_in, const int* in_sizes, int n_in,
                              void* d_out, int out_size, void* d_ws, size_t ws_size,
                              hipStream_t stream) {
  (void)in_sizes; (void)n_in; (void)out_size; (void)ws_size;

  // ---- input indices (setup_inputs dict order, params flattened) ----
  const float* emb  = (const float*)d_in[0];
  const float* msk  = (const float*)d_in[1];
  const int*   wl   = (const int*)d_in[2];
  const float* cw[5] = { (const float*)d_in[3], (const float*)d_in[4], (const float*)d_in[5],
                         (const float*)d_in[6], (const float*)d_in[7] };
  const float* cb[5] = { (const float*)d_in[8], (const float*)d_in[9], (const float*)d_in[10],
                         (const float*)d_in[11], (const float*)d_in[12] };
  const float* aw   = (const float*)d_in[13];
  const float* ab   = (const float*)d_in[14];
  const float* hw_pw[2] = { (const float*)d_in[15], (const float*)d_in[19] };
  const float* hw_pb[2] = { (const float*)d_in[16], (const float*)d_in[20] };
  const float* hw_tw[2] = { (const float*)d_in[17], (const float*)d_in[21] };
  const float* hw_tb[2] = { (const float*)d_in[18], (const float*)d_in[22] };
  const float* ah_g = (const float*)d_in[23];
  const float* ah_b = (const float*)d_in[24];
  const float* ff_w1[2] = { (const float*)d_in[25], (const float*)d_in[31] };
  const float* ff_b1[2] = { (const float*)d_in[26], (const float*)d_in[32] };
  const float* ff_w2[2] = { (const float*)d_in[27], (const float*)d_in[33] };
  const float* ff_b2[2] = { (const float*)d_in[28], (const float*)d_in[34] };
  const float* ff_g[2]  = { (const float*)d_in[29], (const float*)d_in[35] };
  const float* ff_b[2]  = { (const float*)d_in[30], (const float*)d_in[36] };

  // ---- workspace carve-up ----
  char* ws = (char*)d_ws;
  size_t off = 0;
  auto take = [&](size_t bytes) -> void* {
    void* p = ws + off;
    off += (bytes + 255) & ~(size_t)255;
    return p;
  };
  _Float16* x16  = (_Float16*)take((size_t)B_ * LP_ * D_ * 2);  // (B, LP, D) padded
  _Float16* h16  = (_Float16*)take((size_t)B_ * L_ * F_ * 2);   // (B,L,F)
  float*    yA32 = (float*)   take((size_t)B_ * W_ * C_ * 4);   // (B,W,C) ping
  _Float16* yA16 = (_Float16*)take((size_t)B_ * W_ * C_ * 2);
  float*    yB32 = (float*)   take((size_t)B_ * W_ * C_ * 4);   // (B,W,C) pong
  _Float16* yB16 = (_Float16*)take((size_t)B_ * W_ * C_ * 2);
  _Float16* z116 = (_Float16*)take((size_t)BW_ * C2_ * 2);
  float*    z32  = (float*)   take((size_t)BW_ * C_ * 4);
  int*      seg  = (int*)     take((size_t)B_ * L_ * 4);
  const int cwn[5] = { 128 * D_ * 1, 256 * D_ * 3, 512 * D_ * 5, 512 * D_ * 7, 256 * D_ * 9 };
  const int cfc[5] = { 128, 256, 512, 512, 256 };
  const int cks[5] = { 1, 3, 5, 7, 9 };
  _Float16* cw16[5];
  for (int i = 0; i < 5; ++i) cw16[i] = (_Float16*)take((size_t)cwn[i] * 2);
  _Float16* aw16 = (_Float16*)take((size_t)C_ * F_ * 2);
  _Float16* pw16[2], *tw16[2], *w1t16[2], *w2t16[2];
  for (int i = 0; i < 2; ++i) {
    pw16[i]  = (_Float16*)take((size_t)C_ * C_ * 2);
    tw16[i]  = (_Float16*)take((size_t)C_ * C_ * 2);
    w1t16[i] = (_Float16*)take((size_t)C_ * C2_ * 2);  // (C2 x C) transposed
    w2t16[i] = (_Float16*)take((size_t)C2_ * C_ * 2);  // (C x C2) transposed
  }

  auto cvt = [&](_Float16* dst, const float* src, int n) {
    k_f32_to_f16<<<dim3((n + 255) / 256), dim3(256), 0, stream>>>(dst, src, n);
  };
  auto cvtT = [&](_Float16* dst, const float* src, int R, int Cc) {
    int n = R * Cc;
    k_cvt_t<<<dim3((n + 255) / 256), dim3(256), 0, stream>>>(dst, src, R, Cc);
  };

  // ---- weight packing / downconversion ----
  for (int i = 0; i < 5; ++i) {
    k_pack_conv<<<dim3((cwn[i] + 255) / 256), dim3(256), 0, stream>>>(
        cw16[i], cw[i], cfc[i], cks[i]);
  }
  cvt(aw16, aw, C_ * F_);
  for (int i = 0; i < 2; ++i) {
    cvt(pw16[i], hw_pw[i], C_ * C_);
    cvt(tw16[i], hw_tw[i], C_ * C_);
    cvtT(w1t16[i], ff_w1[i], C_, C2_);   // (C,C2) -> (C2,C)
    cvtT(w2t16[i], ff_w2[i], C2_, C_);   // (C2,C) -> (C,C2)
  }

  // ---- input prep (includes zeroed halos) ----
  {
    int n = B_ * LP_ * D_;
    k_prep_x<<<dim3((n + 255) / 256), dim3(256), 0, stream>>>(emb, msk, x16);
  }
  k_seg<<<dim3(B_), dim3(256), 0, stream>>>(wl, seg);
  {
    int n = B_ * W_ * C_;
    k_zero_f32<<<dim3((n + 255) / 256), dim3(256), 0, stream>>>(yA32, n);
  }

  // ---- conv bank (WMMA, unguarded loads via padded x) ----
  {
    int tiles = B_ * (F_ / 16) * (L_ / 32);          // 106496
    k_conv<<<dim3(tiles / 8), dim3(256), 0, stream>>>(
        x16, cw16[0], cw16[1], cw16[2], cw16[3], cw16[4],
        cb[0], cb[1], cb[2], cb[3], cb[4], h16);
  }

  // ---- after_w GEMM + ReLU + segment-max pool ----
  {
    int tiles = B_ * (C_ / 32) * (L_ / 32);          // 16384
    k_after<<<dim3(tiles / 8), dim3(256), 0, stream>>>(h16, aw16, ab, seg, yA32);
  }
  cvt(yA16, yA32, B_ * W_ * C_);

  // ---- highway x2 (ping-pong) ----
  {
    int tiles = B_ * (C_ / 16) * (W_ / 32);          // 8192
    k_highway<<<dim3(tiles / 8), dim3(256), 0, stream>>>(
        pw16[0], hw_pb[0], tw16[0], hw_tb[0], yA32, yA16, yB32, yB16);
    k_highway<<<dim3(tiles / 8), dim3(256), 0, stream>>>(
        pw16[1], hw_pb[1], tw16[1], hw_tb[1], yB32, yB16, yA32, yA16);
  }

  // ---- relu + LN(ah): yA -> yB (both (B,W,C)) ----
  k_relu_ln<<<dim3(B_ * W_), dim3(256), 0, stream>>>(yA32, ah_g, ah_b, yB32, yB16);

  // ---- FF block 0: yB -> yA ----
  {
    int t1 = (BW_ / 32) * (C2_ / 32);                // 8192
    int t2 = (BW_ / 32) * (C_ / 32);                 // 4096
    k_ff1<<<dim3(t1 / 8), dim3(256), 0, stream>>>(yB16, w1t16[0], ff_b1[0], z116);
    k_ff2<<<dim3(t2 / 8), dim3(256), 0, stream>>>(z116, w2t16[0], ff_b2[0], z32);
    k_ln_res<<<dim3(BW_), dim3(256), 0, stream>>>(z32, yB32, ff_g[0], ff_b[0], yA32, yA16);

    // ---- FF block 1: yA -> d_out (f32 only) ----
    k_ff1<<<dim3(t1 / 8), dim3(256), 0, stream>>>(yA16, w1t16[1], ff_b1[1], z116);
    k_ff2<<<dim3(t2 / 8), dim3(256), 0, stream>>>(z116, w2t16[1], ff_b2[1], z32);
    k_ln_res<<<dim3(BW_), dim3(256), 0, stream>>>(z32, yA32, ff_g[1], ff_b[1],
                                                  (float*)d_out, (_Float16*)nullptr);
  }

  // ---- mask tail ----
  {
    float* om = (float*)d_out + (size_t)B_ * W_ * C_;
    int n = B_ * W_;
    k_mask<<<dim3((n + 255) / 256), dim3(256), 0, stream>>>(wl, om);
  }
}
